// MMFP4MoE_27333171871988
// MI455X (gfx1250) — matrix operand order repaired and checked by reading, not compiled, not run on hardware
//
#include <hip/hip_runtime.h>
#include <hip/hip_bf16.h>

// ---------------------------------------------------------------------------
// MoE SwiGLU (E=64, K=4, H=2048, I=1536, C=256, T=2048) for gfx1250 (MI455X).
// Memory-bound on fp32 weight streaming (~2.4GB @ 23.3TB/s ~= 103us); compute
// via v_wmma_f32_16x16x32_bf16 (fp32 accumulate). Weights are streamed once,
// converted to bf16, and staged in LDS in *fragment-major* layout so each
// wave's WMMA operand loads are contiguous ds_load_b128s.
// ---------------------------------------------------------------------------

#define E_   64
#define K_   4
#define H_   2048
#define I_   1536
#define C_   256
#define T_   2048
#define NSH  (T_ / C_)     // 8 shared-expert tiles treated as virtual experts
#define VE   (E_ + NSH)    // 72 virtual experts

typedef __attribute__((ext_vector_type(16))) __bf16 v16bf;
typedef __attribute__((ext_vector_type(8)))  float  v8f;
typedef __attribute__((ext_vector_type(8)))  __bf16 bf16x8;
typedef __attribute__((ext_vector_type(4)))  __bf16 bf16x4;
typedef __attribute__((ext_vector_type(4)))  float  f32x4;

// ---- WMMA fragment loaders (wave32; contiguous k = 16*(lane/16) + j in A and B alike) ----------
// A (16x32, 16-bit), LDS tile As[row][32]:
//   lane<16  -> row=lane,     K halves = {0..15}
//   lane>=16 -> row=lane-16,  K halves = {16..31}
// -> two contiguous 8-half runs => 2x ds_load_b128
__device__ __forceinline__ v16bf frag_A(const __bf16* s, int ld, int row, int lane) {
  v16bf f;
  int kb = (lane < 16) ? 0 : 16;
  const __bf16* p = s + row * ld + kb;
#pragma unroll
  for (int j = 0; j < 8; ++j) f[j] = p[j];
#pragma unroll
  for (int j = 8; j < 16; ++j) f[j] = p[j];
  return f;
}
// B staged fragment-major: BF[lane][16 halves] contiguous => 2x ds_load_b128
__device__ __forceinline__ v16bf frag_Bfm(const __bf16* s, int lane) {
  v16bf f;
  const __bf16* p = s + lane * 16;
#pragma unroll
  for (int j = 0; j < 16; ++j) f[j] = p[j];
  return f;
}

// Stage a 32(K) x NT(N) fp32 weight tile into fragment-major bf16 LDS layout.
// BF layout: [NT/16 n-blocks][32 lanes][16 halves], lane = 16*(k>=16)+(n&15),
// half j = k&15. Thread handles a 4(K)x4(N) patch: coalesced f32x4 global row
// loads, one bf16x4 (8B) LDS store per column.
template <int NT>
__device__ __forceinline__ void stage_B_fm(__bf16 (*BF)[32][16],
                                           const float* __restrict__ gsrc,
                                           long grow_stride, int tid) {
  int kq = (tid & 7) * 4;                // K base: 0,4,...,28
  int nq = (tid >> 3) * (NT / 32);       // N base (NT=128 -> 4 cols/thread)
  constexpr int NC = NT / 32;            // columns per thread
  f32x4 rows[4];
#pragma unroll
  for (int r = 0; r < 4; ++r)
    rows[r] = *(const f32x4*)(gsrc + (long)(kq + r) * grow_stride + nq);
#pragma unroll
  for (int c = 0; c < NC; ++c) {
    int n = nq + c;
    bf16x4 v;
#pragma unroll
    for (int r = 0; r < 4; ++r) v[r] = (__bf16)rows[r][c];
    *(bf16x4*)&BF[n >> 4][((kq & 16) ? 16 : 0) + (n & 15)][kq & 15] = v;
  }
}

__device__ __forceinline__ float silu_f(float g) {
  return g / (1.0f + __expf(-g));
}

__device__ __forceinline__ void atomic_add_f32(float* p, float v) {
  __hip_atomic_fetch_add(p, v, __ATOMIC_RELAXED, __HIP_MEMORY_SCOPE_AGENT);
}

// ---------------------------------------------------------------------------
// K0: zero output + dispatch tables
// ---------------------------------------------------------------------------
__global__ __launch_bounds__(256) void moe_zero_kernel(float* __restrict__ out,
                                                       int* __restrict__ counts,
                                                       int* __restrict__ tok_ec,
                                                       float* __restrict__ w_ec) {
  int i = blockIdx.x * 256 + threadIdx.x;
  if (i < T_ * H_) out[i] = 0.0f;
  if (i < E_) counts[i] = 0;
  if (i < E_ * C_) { tok_ec[i] = 0; w_ec[i] = 0.0f; }
}

// ---------------------------------------------------------------------------
// K1: router logits  (one block per token, 64 threads = one expert each)
// ---------------------------------------------------------------------------
__global__ __launch_bounds__(64) void moe_router_kernel(const float* __restrict__ x,
                                                        const float* __restrict__ w_router,
                                                        float* __restrict__ logits) {
  __shared__ float xs[H_];
  int t = blockIdx.x;
  for (int i = threadIdx.x; i < H_; i += 64) xs[i] = x[(long)t * H_ + i];
  __syncthreads();
  int e = threadIdx.x;
  const float* w = w_router + (long)e * H_;
  float acc = 0.0f;
  for (int h = 0; h < H_; ++h) acc = fmaf(xs[h], w[h], acc);
  logits[t * E_ + e] = acc;
}

// ---------------------------------------------------------------------------
// K2: top-4 + softmax + capacity-slot dispatch (one thread per token)
// ---------------------------------------------------------------------------
__global__ __launch_bounds__(256) void moe_route_kernel(const float* __restrict__ logits,
                                                        int* __restrict__ counts,
                                                        int* __restrict__ tok_ec,
                                                        float* __restrict__ w_ec) {
  int t = blockIdx.x * 256 + threadIdx.x;
  if (t >= T_) return;
  const float* lg = logits + t * E_;
  int idx[K_];
  float val[K_];
  unsigned long long taken = 0ull;
#pragma unroll
  for (int k = 0; k < K_; ++k) {
    float best = -__builtin_inff();
    int bi = 0;
    for (int e = 0; e < E_; ++e) {
      if ((taken >> e) & 1ull) continue;
      float v = lg[e];
      if (v > best) { best = v; bi = e; }
    }
    taken |= 1ull << bi;
    idx[k] = bi;
    val[k] = best;
  }
  float m = val[0];                       // first pick is the max
  float ex[K_], s = 0.0f;
#pragma unroll
  for (int k = 0; k < K_; ++k) { ex[k] = __expf(val[k] - m); s += ex[k]; }
  float inv = 1.0f / s;
#pragma unroll
  for (int k = 0; k < K_; ++k) {
    int e = idx[k];
    int slot = atomicAdd(&counts[e], 1);
    if (slot < C_) {
      tok_ec[e * C_ + slot] = t;
      w_ec[e * C_ + slot] = ex[k] * inv;
    }
  }
}

// ---------------------------------------------------------------------------
// K3: grouped SwiGLU GEMM1  (hm = silu(x@Wg) * (x@Wu), bf16 out)
//   block = 256 threads (8 waves); output tile 64(M) x 128(N); K step 32.
//   Waves arranged 2(M) x 4(N): each wave owns a 32x32 patch ->
//   8 WMMAs per K-step on 2 A-fragment + 4 B-fragment loads.
//   grid: x = (C/64)*(I/128) = 4*12 = 48, y = VE = 72 virtual experts
// ---------------------------------------------------------------------------
__global__ __launch_bounds__(256) void moe_gemm1_kernel(const float* __restrict__ x,
                                                        const float* __restrict__ w_gate,
                                                        const float* __restrict__ w_up,
                                                        const float* __restrict__ w_sh_gate,
                                                        const float* __restrict__ w_sh_up,
                                                        const int* __restrict__ tok_ec,
                                                        __bf16* __restrict__ hm) {
  int ve = blockIdx.y;
  int tile = blockIdx.x;
  int m0 = (tile & 3) * 64;
  int n0 = (tile >> 2) * 128;
  bool sh = (ve >= E_);
  const float* Wg = sh ? w_sh_gate : (w_gate + (long)ve * H_ * I_);
  const float* Wu = sh ? w_sh_up   : (w_up   + (long)ve * H_ * I_);

  __shared__ __bf16 As[64][32];          // row-major A tile
  __shared__ __bf16 BgF[8][32][16];      // fragment-major gate tile (128 N)
  __shared__ __bf16 BuF[8][32][16];      // fragment-major up tile

  int tid = threadIdx.x;
  int lane = tid & 31, wave = tid >> 5;
  int wm = (wave & 1) * 32;              // wave M offset (2 sub-blocks of 16)
  int wn = (wave >> 1) * 32;             // wave N offset (2 sub-blocks of 16)
  int nb0 = wn >> 4;                     // first fragment-major n-block

  // A staging: row = tid/4 (0..63), 8 cols at (tid%4)*8
  int arow = tid >> 2;
  int acol = (tid & 3) * 8;
  int tok = sh ? ((ve - E_) * C_ + m0 + arow) : tok_ec[ve * C_ + m0 + arow];
  const float* xrow = x + (long)tok * H_;

  v8f cg[2][2] = {}, cu[2][2] = {};

  for (int k0 = 0; k0 < H_; k0 += 32) {
    // stage A (float4 loads -> one 16B bf16 store)
    {
      f32x4 a0 = *(const f32x4*)(xrow + k0 + acol);
      f32x4 a1 = *(const f32x4*)(xrow + k0 + acol + 4);
      bf16x8 v;
#pragma unroll
      for (int j = 0; j < 4; ++j) { v[j] = (__bf16)a0[j]; v[j + 4] = (__bf16)a1[j]; }
      *(bf16x8*)&As[arow][acol] = v;
    }
    // stage B tiles in fragment-major layout
    stage_B_fm<128>(BgF, Wg + (long)k0 * I_ + n0, I_, tid);
    stage_B_fm<128>(BuF, Wu + (long)k0 * I_ + n0, I_, tid);
    if (k0 + 32 < H_) {                  // prefetch next weight slab
      __builtin_prefetch((const void*)(Wg + (long)(k0 + 32) * I_ + n0 + (tid & 7) * 4), 0, 1);
      __builtin_prefetch((const void*)(Wu + (long)(k0 + 32) * I_ + n0 + (tid & 7) * 4), 0, 1);
    }
    __syncthreads();

    v16bf a[2], bg[2], bu[2];
#pragma unroll
    for (int mi = 0; mi < 2; ++mi)
      a[mi] = frag_A(&As[0][0], 32, wm + mi * 16 + (lane & 15), lane);
#pragma unroll
    for (int ni = 0; ni < 2; ++ni) {
      bg[ni] = frag_Bfm(&BgF[nb0 + ni][0][0], lane);
      bu[ni] = frag_Bfm(&BuF[nb0 + ni][0][0], lane);
    }
#pragma unroll
    for (int mi = 0; mi < 2; ++mi)
#pragma unroll
      for (int ni = 0; ni < 2; ++ni) {
        cg[mi][ni] = __builtin_amdgcn_wmma_f32_16x16x32_bf16(
            false, a[mi], false, bg[ni], (short)0, cg[mi][ni], false, false);
        cu[mi][ni] = __builtin_amdgcn_wmma_f32_16x16x32_bf16(
            false, a[mi], false, bu[ni], (short)0, cu[mi][ni], false, false);
      }
    __syncthreads();
  }

  // SwiGLU epilogue -> bf16 hm[ve][C][I]
  int N = lane & 15;
  int Moff = (lane < 16) ? 0 : 8;
#pragma unroll
  for (int mi = 0; mi < 2; ++mi)
#pragma unroll
    for (int ni = 0; ni < 2; ++ni) {
      long col = n0 + wn + ni * 16 + N;
#pragma unroll
      for (int r = 0; r < 8; ++r) {
        int row = m0 + wm + mi * 16 + Moff + r;
        hm[((long)ve * C_ + row) * I_ + col] =
            (__bf16)(silu_f(cg[mi][ni][r]) * cu[mi][ni][r]);
      }
    }
}

// ---------------------------------------------------------------------------
// K4: GEMM2 (y = hm @ Wd), scale by combine weight, atomic scatter-add to out
//   tile 64(M) x 128(N), K over I in steps of 32; waves 2(M) x 4(N).
//   grid: x = (C/64)*(H/128) = 4*16 = 64, y = VE = 72
// ---------------------------------------------------------------------------
__global__ __launch_bounds__(256) void moe_gemm2_kernel(const __bf16* __restrict__ hm,
                                                        const float* __restrict__ w_down,
                                                        const float* __restrict__ w_sh_down,
                                                        const int* __restrict__ tok_ec,
                                                        const float* __restrict__ w_ec,
                                                        float* __restrict__ out) {
  int ve = blockIdx.y;
  int tile = blockIdx.x;
  int m0 = (tile & 3) * 64;
  int n0 = (tile >> 2) * 128;
  bool sh = (ve >= E_);
  const float* Wd = sh ? w_sh_down : (w_down + (long)ve * I_ * H_);

  __shared__ __bf16 As[64][32];
  __shared__ __bf16 BdF[8][32][16];

  int tid = threadIdx.x;
  int lane = tid & 31, wave = tid >> 5;
  int wm = (wave & 1) * 32;
  int wn = (wave >> 1) * 32;
  int nb0 = wn >> 4;

  int arow = tid >> 2;
  int acol = (tid & 3) * 8;
  const __bf16* hrow = hm + ((long)ve * C_ + m0 + arow) * I_;

  v8f c[2][2] = {};

  for (int k0 = 0; k0 < I_; k0 += 32) {
    *(bf16x8*)&As[arow][acol] = *(const bf16x8*)(hrow + k0 + acol);
    stage_B_fm<128>(BdF, Wd + (long)k0 * H_ + n0, H_, tid);
    if (k0 + 32 < I_)
      __builtin_prefetch((const void*)(Wd + (long)(k0 + 32) * H_ + n0 + (tid & 7) * 4), 0, 1);
    __syncthreads();

    v16bf a[2], b[2];
#pragma unroll
    for (int mi = 0; mi < 2; ++mi)
      a[mi] = frag_A(&As[0][0], 32, wm + mi * 16 + (lane & 15), lane);
#pragma unroll
    for (int ni = 0; ni < 2; ++ni)
      b[ni] = frag_Bfm(&BdF[nb0 + ni][0][0], lane);
#pragma unroll
    for (int mi = 0; mi < 2; ++mi)
#pragma unroll
      for (int ni = 0; ni < 2; ++ni)
        c[mi][ni] = __builtin_amdgcn_wmma_f32_16x16x32_bf16(
            false, a[mi], false, b[ni], (short)0, c[mi][ni], false, false);
    __syncthreads();
  }

  // weighted scatter-add combine
  int N = lane & 15;
  int Moff = (lane < 16) ? 0 : 8;
#pragma unroll
  for (int mi = 0; mi < 2; ++mi) {
#pragma unroll
    for (int r = 0; r < 8; ++r) {
      int row = m0 + wm + mi * 16 + Moff + r;
      float w;
      int tok;
      if (sh) { w = 1.0f; tok = (ve - E_) * C_ + row; }
      else    { w = w_ec[ve * C_ + row]; tok = tok_ec[ve * C_ + row]; }
      if (w != 0.0f) {
        float* o = out + (long)tok * H_ + n0 + wn;
#pragma unroll
        for (int ni = 0; ni < 2; ++ni)
          atomic_add_f32(o + ni * 16 + N, c[mi][ni][r] * w);
      }
    }
  }
}

// ---------------------------------------------------------------------------
extern "C" void kernel_launch(void* const* d_in, const int* in_sizes, int n_in,
                              void* d_out, int out_size, void* d_ws, size_t ws_size,
                              hipStream_t stream) {
  const float* x         = (const float*)d_in[0];
  const float* w_router  = (const float*)d_in[1];
  const float* w_gate    = (const float*)d_in[2];
  const float* w_up      = (const float*)d_in[3];
  const float* w_down    = (const float*)d_in[4];
  const float* w_sh_gate = (const float*)d_in[5];
  const float* w_sh_up   = (const float*)d_in[6];
  const float* w_sh_down = (const float*)d_in[7];
  float* out = (float*)d_out;

  // workspace layout (~57.3 MB total)
  float*  logits = (float*)d_ws;                       // T*E
  int*    counts = (int*)(logits + (long)T_ * E_);     // E
  int*    tok_ec = counts + E_;                        // E*C
  float*  w_ec   = (float*)(tok_ec + E_ * C_);         // E*C
  __bf16* hm     = (__bf16*)(w_ec + E_ * C_);          // VE*C*I bf16

  (void)in_sizes; (void)n_in; (void)out_size; (void)ws_size;

  moe_zero_kernel<<<(T_ * H_ + 255) / 256, 256, 0, stream>>>(out, counts, tok_ec, w_ec);
  moe_router_kernel<<<T_, 64, 0, stream>>>(x, w_router, logits);
  moe_route_kernel<<<(T_ + 255) / 256, 256, 0, stream>>>(logits, counts, tok_ec, w_ec);
  moe_gemm1_kernel<<<dim3((C_ / 64) * (I_ / 128), VE), 256, 0, stream>>>(
      x, w_gate, w_up, w_sh_gate, w_sh_up, tok_ec, hm);
  moe_gemm2_kernel<<<dim3((C_ / 64) * (H_ / 128), VE), 256, 0, stream>>>(
      hm, w_down, w_sh_down, tok_ec, w_ec, out);
}
